// ImplFuncAttention_7198365188578
// MI455X (gfx1250) — compile-verified, matrix operand order checked by reading, not run on hardware
//
#include <hip/hip_runtime.h>
#include <hip/hip_bf16.h>
#include <math.h>

typedef __attribute__((ext_vector_type(16))) _Float16 v16h;
typedef __attribute__((ext_vector_type(8)))  float    v8f;

#define B_   4
#define N_   8448
#define C_   512
#define H_   8
#define HD_  64
#define NL_  256
#define NP_  8192
#define M_   (B_ * N_)          /* 33792 */
#define QKVN (3 * C_)           /* 1536  */

#define WMMA(c, a, b) \
  __builtin_amdgcn_wmma_f32_16x16x32_f16(false, (a), false, (b), (short)0, (c), false, false)

// Async DMA: 16 bytes per lane, global -> LDS, tracked by ASYNCcnt.
__device__ __forceinline__ void async_ld_b128(unsigned lds_off, const void* g) {
  asm volatile("global_load_async_to_lds_b128 %0, %1, off"
               :: "v"(lds_off), "v"(g) : "memory");
}
__device__ __forceinline__ void wait_async0() {
  asm volatile("s_wait_asynccnt 0" ::: "memory");
}

// A/B fragment for 16x32 (f16) tile stored [laneIdx][k] in LDS.
// lane&15 selects the row (A) / column (B); K mapping per ISA 7.12.2.
__device__ __forceinline__ v16h frag_ld(const _Float16* __restrict__ base, int stride, int lane) {
  v16h f;
  const int r = lane & 15, hh = lane >> 4;
#pragma unroll
  for (int j = 0; j < 8; ++j) {
    int k = ((j < 4) ? (2 * j) : (2 * j + 8)) + 8 * hh;
    f[2 * j]     = base[r * stride + k];
    f[2 * j + 1] = base[r * stride + k + 1];
  }
  return f;
}

// B fragment when LDS tile is stored [k][n] (row-major K).
__device__ __forceinline__ v16h frag_ld_T(const _Float16* __restrict__ base, int stride, int lane) {
  v16h f;
  const int n = lane & 15, hh = lane >> 4;
#pragma unroll
  for (int j = 0; j < 8; ++j) {
    int k = ((j < 4) ? (2 * j) : (2 * j + 8)) + 8 * hh;
    f[2 * j]     = base[k * stride + n];
    f[2 * j + 1] = base[(k + 1) * stride + n];
  }
  return f;
}

// ---------------------------------------------------------------------------
// Kernel 1: qkv = x @ w_qkv^T, scattered into per-head f16 q/k/v buffers.
// Block: 256 thr (8 waves), tile 128x128, K-chunks of 32.
// ---------------------------------------------------------------------------
__global__ __launch_bounds__(256) void qkv_gemm(
    const float* __restrict__ x, const float* __restrict__ wqkv,
    _Float16* __restrict__ qB, _Float16* __restrict__ kB, _Float16* __restrict__ vB) {
  __shared__ _Float16 sA[128][40];
  __shared__ _Float16 sB[128][40];
  const int tid = threadIdx.x, lane = tid & 31, wid = tid >> 5;
  const int m0 = blockIdx.x * 128, n0 = blockIdx.y * 128;
  const int wr = wid >> 1, wc = wid & 1;   // 4x2 wave grid, wave tile 32x64
  const int hh = lane >> 4, ln = lane & 15;

  v8f zero = {};
  v8f acc[2][4];
#pragma unroll
  for (int i = 0; i < 2; ++i)
#pragma unroll
    for (int j = 0; j < 4; ++j) acc[i][j] = zero;

  for (int k0 = 0; k0 < C_; k0 += 32) {
    if (k0 + 32 < C_) {  // prefetch next K tile (global_prefetch_b8)
      __builtin_prefetch(&x[(size_t)(m0 + (tid >> 1)) * C_ + k0 + 32 + (tid & 1) * 16], 0, 0);
      __builtin_prefetch(&wqkv[(size_t)(n0 + (tid >> 1)) * C_ + k0 + 32 + (tid & 1) * 16], 0, 0);
    }
    for (int idx = tid; idx < 128 * 32; idx += 256) {
      int r = idx >> 5, c = idx & 31;
      sA[r][c] = (_Float16)x[(size_t)(m0 + r) * C_ + k0 + c];
      sB[r][c] = (_Float16)wqkv[(size_t)(n0 + r) * C_ + k0 + c];
    }
    __syncthreads();
#pragma unroll
    for (int mt = 0; mt < 2; ++mt) {
      v16h a = frag_ld(&sA[wr * 32 + mt * 16][0], 40, lane);
#pragma unroll
      for (int nt = 0; nt < 4; ++nt) {
        v16h b = frag_ld(&sB[wc * 64 + nt * 16][0], 40, lane);
        acc[mt][nt] = WMMA(acc[mt][nt], a, b);
      }
    }
    __syncthreads();
  }

#pragma unroll
  for (int mt = 0; mt < 2; ++mt)
#pragma unroll
    for (int nt = 0; nt < 4; ++nt)
#pragma unroll
      for (int r = 0; r < 8; ++r) {
        int m  = m0 + wr * 32 + mt * 16 + r + 8 * hh;
        int nn = n0 + wc * 64 + nt * 16 + ln;
        int s = nn >> 9, rem = nn & 511, h = rem >> 6, d = rem & 63;
        int bb = m / N_, tok = m - bb * N_;
        _Float16* dst = (s == 0) ? qB : (s == 1) ? kB : vB;
        dst[((size_t)((bb * H_ + h) * N_) + tok) * HD_ + d] = (_Float16)acc[mt][nt][r];
      }
}

// ---------------------------------------------------------------------------
// Kernel 2: attention. mode==0: latent self-attn rows; mode==1: point rows
// (cross + per-row self slot, attn_mean accumulation).
// Block handles one (b, h, 64-row tile). Full 256x64 K/V resident in LDS,
// staged via ASYNCcnt-tracked global_load_async_to_lds_b128.
// ---------------------------------------------------------------------------
__global__ __launch_bounds__(256) void attn_kernel(
    const _Float16* __restrict__ qB, const _Float16* __restrict__ kB,
    const _Float16* __restrict__ vB, _Float16* __restrict__ cOut,
    float* __restrict__ meanOut, int mode, int nTiles) {
  __shared__ _Float16 sK[NL_][72];
  __shared__ _Float16 sV[NL_][72];
  __shared__ _Float16 sQ[64][72];
  __shared__ _Float16 sP[64][264];
  __shared__ float sRed[64][4];
  __shared__ float sMax[64], sInv[64], sSelfS[64], sSelfA[64];

  int tmp = blockIdx.x;
  const int rowTile = tmp % nTiles; tmp /= nTiles;
  const int h = tmp % H_, b = tmp / H_;
  const int tid = threadIdx.x, lane = tid & 31, wid = tid >> 5;
  const int hh = lane >> 4, ln = lane & 15;
  const int row0 = rowTile * 64;
  const float scale = 0.125f;   // hd^-0.5, hd=64
  const size_t headBase = (size_t)((b * H_ + h) * N_) * HD_;

  // ---- async DMA K_l, V_l (256x64 f16 each) and Q tile (64x64 f16) to LDS
  {
    const _Float16* kSrc = kB + headBase;
    const _Float16* vSrc = vB + headBase;
    const int qtok0 = mode ? (NL_ + row0) : row0;
    const _Float16* qSrc = qB + headBase + (size_t)qtok0 * HD_;
#pragma unroll
    for (int it = 0; it < 8; ++it) {           // 2048 16B chunks per matrix
      int chunk = tid + it * 256;
      int l = chunk >> 3, c = (chunk & 7) * 8; // row l, 8-half chunk c
      async_ld_b128((unsigned)(size_t)&sK[l][c], kSrc + chunk * 8);
      async_ld_b128((unsigned)(size_t)&sV[l][c], vSrc + chunk * 8);
    }
#pragma unroll
    for (int it = 0; it < 2; ++it) {           // 512 16B chunks
      int chunk = tid + it * 256;
      int r = chunk >> 3, c = (chunk & 7) * 8;
      async_ld_b128((unsigned)(size_t)&sQ[r][c], qSrc + chunk * 8);
    }
    wait_async0();
  }
  __syncthreads();

  if (mode) {  // per-row self score q_p . k_p
    int r = tid >> 2, part = tid & 3;
    int tok = NL_ + row0 + r;
    float s = 0.f;
    for (int d = part * 16; d < part * 16 + 16; ++d)
      s += (float)sQ[r][d] * (float)kB[headBase + (size_t)tok * HD_ + d];
    sRed[r][part] = s;
    __syncthreads();
    if (tid < 64)
      sSelfS[tid] = scale * (sRed[tid][0] + sRed[tid][1] + sRed[tid][2] + sRed[tid][3]);
    __syncthreads();
  }

  // ---- scores = scale * Q Kl^T  (64x256), 8 waves: 16 rows x 128 cols each
  {
    const int rt = wid >> 1, ch = wid & 1;
    v8f zero = {};
    v8f acc[8];
#pragma unroll
    for (int i = 0; i < 8; ++i) acc[i] = zero;
#pragma unroll
    for (int kk = 0; kk < HD_; kk += 32) {
      v16h a = frag_ld(&sQ[rt * 16][kk], 72, lane);
#pragma unroll
      for (int nt = 0; nt < 8; ++nt) {
        v16h bf = frag_ld(&sK[ch * 128 + nt * 16][kk], 72, lane);
        acc[nt] = WMMA(acc[nt], a, bf);
      }
    }
#pragma unroll
    for (int nt = 0; nt < 8; ++nt)
#pragma unroll
      for (int r = 0; r < 8; ++r) {
        int mr = rt * 16 + r + 8 * hh;
        int nc = ch * 128 + nt * 16 + ln;
        sP[mr][nc] = (_Float16)(acc[nt][r] * scale);
      }
  }
  __syncthreads();

  // ---- softmax over 256 (+1 self when mode), in place on sP
  {
    const int r = tid >> 2, part = tid & 3;
    float mx = -1e30f;
    for (int c = part * 64; c < part * 64 + 64; ++c) mx = fmaxf(mx, (float)sP[r][c]);
    sRed[r][part] = mx;
    __syncthreads();
    if (tid < 64) {
      float m4 = fmaxf(fmaxf(sRed[tid][0], sRed[tid][1]), fmaxf(sRed[tid][2], sRed[tid][3]));
      if (mode) m4 = fmaxf(m4, sSelfS[tid]);
      sMax[tid] = m4;
    }
    __syncthreads();
    const float rm = sMax[r];
    float sm = 0.f;
    for (int c = part * 64; c < part * 64 + 64; ++c) sm += __expf((float)sP[r][c] - rm);
    sRed[r][part] = sm;
    __syncthreads();
    if (tid < 64) {
      float tot = sRed[tid][0] + sRed[tid][1] + sRed[tid][2] + sRed[tid][3];
      float inv;
      if (mode) {
        float es = __expf(sSelfS[tid] - sMax[tid]);
        inv = 1.f / (tot + es);
        sSelfA[tid] = es * inv;
      } else {
        inv = 1.f / tot;
      }
      sInv[tid] = inv;
    }
    __syncthreads();
    const float inv = sInv[r];
    for (int c = part * 64; c < part * 64 + 64; ++c) {
      float p = __expf((float)sP[r][c] - rm) * inv;
      sP[r][c] = (_Float16)p;
      if (mode)  // attn_mean = mean over heads of cross part
        atomicAdd(&meanOut[((size_t)(b * NP_ + row0 + r)) * NL_ + c], p * 0.125f);
    }
  }
  __syncthreads();

  // ---- out = P @ Vl (64x64), 8 waves: 16 rows x 32 cols each; add self term
  {
    const int rt = wid >> 1, cb = (wid & 1) * 2;
    v8f acc0 = {}, acc1 = {};
#pragma unroll
    for (int kk = 0; kk < NL_; kk += 32) {
      v16h a  = frag_ld(&sP[rt * 16][kk], 264, lane);
      v16h b0 = frag_ld_T(&sV[kk][cb * 16], 72, lane);
      v16h b1 = frag_ld_T(&sV[kk][(cb + 1) * 16], 72, lane);
      acc0 = WMMA(acc0, a, b0);
      acc1 = WMMA(acc1, a, b1);
    }
#pragma unroll
    for (int j = 0; j < 2; ++j) {
      v8f acc = j ? acc1 : acc0;
      int ct = cb + j;
#pragma unroll
      for (int r = 0; r < 8; ++r) {
        int mr = rt * 16 + r + 8 * hh;
        int d  = ct * 16 + ln;
        float val = acc[r];
        int tok;
        if (mode) {
          tok = NL_ + row0 + mr;
          val += sSelfA[mr] * (float)vB[headBase + (size_t)tok * HD_ + d];
        } else {
          tok = row0 + mr;
        }
        cOut[((size_t)(b * N_ + tok)) * C_ + h * HD_ + d] = (_Float16)val;
      }
    }
  }
}

// ---------------------------------------------------------------------------
// Kernel 3: out = concat @ w_proj^T + b_proj  (f32 out)
// ---------------------------------------------------------------------------
__global__ __launch_bounds__(256) void proj_gemm(
    const _Float16* __restrict__ cIn, const float* __restrict__ wproj,
    const float* __restrict__ bias, float* __restrict__ out) {
  __shared__ _Float16 sA[128][40];
  __shared__ _Float16 sB[128][40];
  const int tid = threadIdx.x, lane = tid & 31, wid = tid >> 5;
  const int m0 = blockIdx.x * 128, n0 = blockIdx.y * 128;
  const int wr = wid >> 1, wc = wid & 1;
  const int hh = lane >> 4, ln = lane & 15;

  v8f zero = {};
  v8f acc[2][4];
#pragma unroll
  for (int i = 0; i < 2; ++i)
#pragma unroll
    for (int j = 0; j < 4; ++j) acc[i][j] = zero;

  for (int k0 = 0; k0 < C_; k0 += 32) {
    if (k0 + 32 < C_) {
      __builtin_prefetch(&cIn[(size_t)(m0 + (tid >> 1)) * C_ + k0 + 32 + (tid & 1) * 16], 0, 0);
      __builtin_prefetch(&wproj[(size_t)(n0 + (tid >> 1)) * C_ + k0 + 32 + (tid & 1) * 16], 0, 0);
    }
    for (int idx = tid; idx < 128 * 32; idx += 256) {
      int r = idx >> 5, c = idx & 31;
      sA[r][c] = cIn[(size_t)(m0 + r) * C_ + k0 + c];
      sB[r][c] = (_Float16)wproj[(size_t)(n0 + r) * C_ + k0 + c];
    }
    __syncthreads();
#pragma unroll
    for (int mt = 0; mt < 2; ++mt) {
      v16h a = frag_ld(&sA[wr * 32 + mt * 16][0], 40, lane);
#pragma unroll
      for (int nt = 0; nt < 4; ++nt) {
        v16h b = frag_ld(&sB[wc * 64 + nt * 16][0], 40, lane);
        acc[mt][nt] = WMMA(acc[mt][nt], a, b);
      }
    }
    __syncthreads();
  }

#pragma unroll
  for (int mt = 0; mt < 2; ++mt)
#pragma unroll
    for (int nt = 0; nt < 4; ++nt)
#pragma unroll
      for (int r = 0; r < 8; ++r) {
        int m  = m0 + wr * 32 + mt * 16 + r + 8 * hh;
        int nn = n0 + wc * 64 + nt * 16 + ln;
        out[(size_t)m * C_ + nn] = acc[mt][nt][r] + bias[nn];
      }
}

__global__ void zero_f32(float* __restrict__ p, size_t n) {
  size_t i = (size_t)blockIdx.x * blockDim.x + threadIdx.x;
  size_t stride = (size_t)gridDim.x * blockDim.x;
  for (; i < n; i += stride) p[i] = 0.f;
}

extern "C" void kernel_launch(void* const* d_in, const int* in_sizes, int n_in,
                              void* d_out, int out_size, void* d_ws, size_t ws_size,
                              hipStream_t stream) {
  (void)in_sizes; (void)n_in; (void)out_size; (void)ws_size;
  const float* x     = (const float*)d_in[0];
  const float* wqkv  = (const float*)d_in[1];
  const float* wproj = (const float*)d_in[2];
  const float* bias  = (const float*)d_in[3];
  float* out     = (float*)d_out;
  float* meanOut = out + (size_t)M_ * C_;   // attn_mean region of d_out

  char* ws = (char*)d_ws;
  const size_t perBuf = (size_t)B_ * H_ * N_ * HD_ * sizeof(_Float16);  // 34.6 MB
  _Float16* qB = (_Float16*)(ws);
  _Float16* kB = (_Float16*)(ws + perBuf);
  _Float16* vB = (_Float16*)(ws + 2 * perBuf);
  _Float16* cB = (_Float16*)(ws + 3 * perBuf);  // concat attention output, f16

  zero_f32<<<2048, 256, 0, stream>>>(meanOut, (size_t)B_ * NP_ * NL_);
  qkv_gemm<<<dim3(M_ / 128, QKVN / 128), 256, 0, stream>>>(x, wqkv, qB, kB, vB);
  attn_kernel<<<B_ * H_ * (NL_ / 64), 256, 0, stream>>>(qB, kB, vB, cB, meanOut, 0, NL_ / 64);
  attn_kernel<<<B_ * H_ * (NP_ / 64), 256, 0, stream>>>(qB, kB, vB, cB, meanOut, 1, NP_ / 64);
  proj_gemm<<<dim3(M_ / 128, C_ / 128), 256, 0, stream>>>(cB, wproj, bias, out);
}
